// Unet_90134183673998
// MI455X (gfx1250) — matrix operand order repaired and checked by reading, not compiled, not run on hardware
//
#include <hip/hip_runtime.h>
#include <hip/hip_bf16.h>
#include <math.h>

// ---------------------------------------------------------------------------
// CDNA5 (gfx1250) wave32 WMMA types
// ---------------------------------------------------------------------------
typedef __attribute__((ext_vector_type(16))) _Float16 v16h;
typedef __attribute__((ext_vector_type(8)))  float    v8f;

#define BATCH    8
#define HW0      256
#define FLATLEN  9490

// --- WMMA 16x16x32 f16 operand k order used here (contiguous k = 16*half + h, wave32) ----------
// A (16x32): lanes 0-15 hold M=lane, K=0..15; lanes 16-31: K=16..31
__device__ __forceinline__ int a_klocal(int lane, int h) {
    return ((lane >= 16) ? 16 : 0) + h;
}
// B (32x16): lanes 0-15 hold N=lane, K=0..15; lanes 16-31: K=16..31
__device__ __forceinline__ int b_klocal(int lane, int h) {
    return ((lane >= 16) ? 16 : 0) + h;
}

// Unconditional vectorized A-operand load from padded f32 row (stride >= k0+32)
__device__ __forceinline__ v16h load_a_fast(const float* __restrict__ row,
                                            int k0, int lane) {
    const int base = (lane >= 16) ? 16 : 0;
    const float4* p0 = (const float4*)(row + k0 + base);        // 16B aligned
    const float4* p1 = (const float4*)(row + k0 + base + 8);
    float4 f0 = p0[0], f1 = p0[1], f2 = p1[0], f3 = p1[1];
    v16h a;
    a[0]=(_Float16)f0.x; a[1]=(_Float16)f0.y; a[2]=(_Float16)f0.z; a[3]=(_Float16)f0.w;
    a[4]=(_Float16)f1.x; a[5]=(_Float16)f1.y; a[6]=(_Float16)f1.z; a[7]=(_Float16)f1.w;
    a[8]=(_Float16)f2.x; a[9]=(_Float16)f2.y; a[10]=(_Float16)f2.z; a[11]=(_Float16)f2.w;
    a[12]=(_Float16)f3.x; a[13]=(_Float16)f3.y; a[14]=(_Float16)f3.z; a[15]=(_Float16)f3.w;
    return a;
}
// Unconditional vectorized B-operand load: 16 contiguous f32 at row+koff
__device__ __forceinline__ v16h load_b_fast(const float* __restrict__ row,
                                            int koff) {
    const float4* p = (const float4*)(row + koff);              // 16B aligned
    float4 f0 = p[0], f1 = p[1], f2 = p[2], f3 = p[3];
    v16h b;
    b[0]=(_Float16)f0.x; b[1]=(_Float16)f0.y; b[2]=(_Float16)f0.z; b[3]=(_Float16)f0.w;
    b[4]=(_Float16)f1.x; b[5]=(_Float16)f1.y; b[6]=(_Float16)f1.z; b[7]=(_Float16)f1.w;
    b[8]=(_Float16)f2.x; b[9]=(_Float16)f2.y; b[10]=(_Float16)f2.z; b[11]=(_Float16)f2.w;
    b[12]=(_Float16)f3.x; b[13]=(_Float16)f3.y; b[14]=(_Float16)f3.z; b[15]=(_Float16)f3.w;
    return b;
}

// ---------------------------------------------------------------------------
// Hypernetwork skinny GEMM: out[b,n] = sum_k h[b,k] * W[n,k]
// M=16 (8 batch rows + 8 zero-padded rows in h_in), one wave per 16-wide N
// tile, K split across gridDim.y into disjoint partials. h_in is padded to
// [16][lda] (lda = K rounded up to 32, pad pre-zeroed) so A loads are
// unconditional float4s. W streamed once: fast path = 4x float4, single
// clamp+select tail step. Memory-bound by design (~1.2 GB of fp32 weights).
// ---------------------------------------------------------------------------
__global__ void __launch_bounds__(128)
hyper_gemm_wmma(const float* __restrict__ h_in, int lda,
                const float* __restrict__ Wt,
                float* __restrict__ partial, int K, int N) {
    const int lane  = threadIdx.x & 31;
    const int wave  = threadIdx.x >> 5;
    const int ntile = blockIdx.x * 4 + wave;
    const int n0    = ntile * 16;
    if (n0 >= N) return;                       // wave-uniform

    const int S   = (K + 31) / 32;             // total 32-wide K steps
    const int per = (S + gridDim.y - 1) / gridDim.y;
    const int s0  = blockIdx.y * per;
    int       s1  = s0 + per; if (s1 > S) s1 = S;
    const int s_fast = (K / 32 < s1) ? (K / 32) : s1;   // steps with k0+32<=K

    const int  m    = lane & 15;
    const int  ncol = n0 + (lane & 15);
    const bool nvalid = (ncol < N);
    const float* wrow = Wt + (size_t)(nvalid ? ncol : 0) * (size_t)K;
    const float* arow = h_in + (size_t)m * lda;
    const int  bbase = (lane >= 16) ? 16 : 0;

    v8f acc = {};
    int s = s0;
    #pragma unroll 2
    for (; s < s_fast; ++s) {
        const int k0 = s * 32;
        v16h a = load_a_fast(arow, k0, lane);
        v16h b = load_b_fast(wrow, k0 + bbase);
        __builtin_prefetch(wrow + k0 + bbase + 64, 0, 0);   // global_prefetch_b8
        acc = __builtin_amdgcn_wmma_f32_16x16x32_f16(
            false, a, false, b, (short)0, acc, false, false);
    }
    for (; s < s1; ++s) {                      // at most one tail step
        const int k0 = s * 32;
        v16h a = load_a_fast(arow, k0, lane);  // pad region is zeroed
        v16h b;
        #pragma unroll
        for (int h = 0; h < 16; ++h) {
            int   kk = k0 + bbase + h;
            int   kc = (kk < K) ? kk : (K - 1);       // clamp: load always valid
            float v  = wrow[kc];
            b[h] = (_Float16)((kk < K) ? v : 0.0f);   // select, no branch
        }
        acc = __builtin_amdgcn_wmma_f32_16x16x32_f16(
            false, a, false, b, (short)0, acc, false, false);
    }
    // C/D layout: VGPR r, lanes 0-15 -> M=r (rows 0..7 == batch), N=lane
    if (lane < 16) {
        const int n = n0 + lane;
        if (n < N) {
            #pragma unroll
            for (int r = 0; r < BATCH; ++r)
                partial[((size_t)blockIdx.y * BATCH + r) * (size_t)N + n] = acc[r];
        }
    }
}

// Reduce K-split partials, add bias, optional leaky-ReLU; writes the FULL
// padded [rows][ldo] tile (zeros outside b<8, n<N) so downstream A loads
// need no guards. For the final (flat) layer rows=8, ldo=N -> compact.
__global__ void bias_act_kernel(const float* __restrict__ partial, int nsplit,
                                const float* __restrict__ bias,
                                float* __restrict__ out, int N, int ldo,
                                int rows, int act) {
    int i = blockIdx.x * blockDim.x + threadIdx.x;
    if (i >= rows * ldo) return;
    int b = i / ldo, n = i - b * ldo;
    float s = 0.0f;
    if (b < BATCH && n < N) {
        for (int ks = 0; ks < nsplit; ++ks)
            s += partial[((size_t)ks * BATCH + b) * (size_t)N + n];
        s += bias[n];
        if (act) s = (s > 0.0f) ? s : 0.01f * s;
    }
    out[i] = s;
}

// First MLP layer (K=1): writes padded [16][128] tile, zeros in pad region
__global__ void h1_kernel(const float* __restrict__ hyper,
                          const float* __restrict__ w1,
                          const float* __restrict__ b1,
                          float* __restrict__ h1) {
    int i = blockIdx.x * blockDim.x + threadIdx.x;
    if (i >= 16 * 128) return;
    int b = i >> 7, n = i & 127;
    float v = 0.0f;
    if (b < BATCH && n < 100) {
        v = hyper[b] * w1[n] + b1[n];
        v = (v > 0.0f) ? v : 0.01f * v;
    }
    h1[i] = v;
}

// L1 over per-sample generated parameters
__global__ void l1_kernel(const float* __restrict__ flat, float* __restrict__ out_l1) {
    __shared__ float red[8];
    const int b = blockIdx.x;
    float s = 0.0f;
    for (int i = threadIdx.x; i < FLATLEN; i += blockDim.x)
        s += fabsf(flat[(size_t)b * FLATLEN + i]);
    for (int off = 16; off > 0; off >>= 1) s += __shfl_down(s, off, 32);
    if ((threadIdx.x & 31) == 0) red[threadIdx.x >> 5] = s;
    __syncthreads();
    if (threadIdx.x == 0) {
        float t = 0.0f;
        for (int w = 0; w < (int)(blockDim.x >> 5); ++w) t += red[w];
        out_l1[b] = t;
    }
}

// ---------------------------------------------------------------------------
// Implicit-GEMM conv via WMMA: M=16 row pixels, N=16 (Cout padded),
// K = CIN*KS*KS (compile-time). All loads unconditional (clamp + select),
// S = ceil(K/32) in {1,3,5} fully unrolls -> straight-line WMMA chain.
// ---------------------------------------------------------------------------
template <int KS, int CIN>
__global__ void __launch_bounds__(32)
conv_wmma(const float* __restrict__ x, const float* __restrict__ flat,
          float* __restrict__ y, int Cout, int H, int W,
          int w_off, int b_off, int relu) {
    constexpr int K   = CIN * KS * KS;
    constexpr int S   = (K + 31) / 32;
    constexpr int PAD = KS / 2;
    const int lane = threadIdx.x & 31;
    const int b    = blockIdx.z;
    const int y0   = blockIdx.y;
    const int x0   = blockIdx.x * 16;
    const float* fw = flat + (size_t)b * FLATLEN + w_off;
    const float* xb = x + (size_t)b * CIN * H * W;

    const int m   = lane & 15;       // pixel within tile (A rows); x0+m < W always
    const int px  = x0 + m;
    const int nco = lane & 15;       // output channel (B cols)

    v8f acc = {};
    #pragma unroll
    for (int s = 0; s < S; ++s) {
        const int k0 = s * 32;
        v16h a, bm;
        // A: im2col gather; clamp coords into the image, select halo/K-pad to 0
        #pragma unroll
        for (int h = 0; h < 16; ++h) {
            const int kk  = k0 + a_klocal(lane, h);
            const int kc  = (kk < K) ? kk : (K - 1);
            const int ci  = kc / (KS * KS);
            const int r   = kc - ci * (KS * KS);
            const int dy  = r / KS - PAD;
            const int dx  = r - (r / KS) * KS - PAD;
            int yy = y0 + dy, xx = px + dx;
            const bool inb = (kk < K) & (yy >= 0) & (yy < H) & (xx >= 0) & (xx < W);
            yy = (yy < 0) ? 0 : ((yy >= H) ? H - 1 : yy);
            xx = (xx < 0) ? 0 : ((xx >= W) ? W - 1 : xx);
            float v = xb[((size_t)ci * H + yy) * W + xx];     // always in-bounds
            a[h] = (_Float16)(inb ? v : 0.0f);
        }
        // B: generated weights w[co][ci][ky][kx] == fw[co*K + kk]; clamp+select
        #pragma unroll
        for (int h = 0; h < 16; ++h) {
            const int  kk    = k0 + b_klocal(lane, h);
            const bool valid = (nco < Cout) & (kk < K);
            const int  idx   = valid ? (nco * K + kk) : 0;
            float v = fw[idx];
            bm[h] = (_Float16)(valid ? v : 0.0f);
        }
        acc = __builtin_amdgcn_wmma_f32_16x16x32_f16(
            false, a, false, bm, (short)0, acc, false, false);
    }
    // D: VGPR r, lanes 0-15 -> pixel r, lanes 16-31 -> pixel 8+r; N = lane&15
    const int co = lane & 15;
    if (co < Cout) {
        const float bias = flat[(size_t)b * FLATLEN + b_off + co];
        #pragma unroll
        for (int r = 0; r < 8; ++r) {
            const int mm  = r + ((lane >= 16) ? 8 : 0);
            const int pxx = x0 + mm;                 // always < W
            float v = acc[r] + bias;
            if (relu && v < 0.0f) v = 0.0f;
            y[(((size_t)b * Cout + co) * H + y0) * W + pxx] = v;
        }
    }
}

// ---------------------------------------------------------------------------
// Elementwise / data-movement kernels
// ---------------------------------------------------------------------------
__global__ void nhwc_to_nchw(const float* __restrict__ zf, float* __restrict__ x) {
    int i = blockIdx.x * blockDim.x + threadIdx.x;
    if (i >= BATCH * 2 * HW0 * HW0) return;
    int xx = i % HW0; int t = i / HW0;
    int yy = t % HW0; t /= HW0;
    int c  = t % 2;   int b = t / 2;
    x[i] = zf[(((size_t)b * HW0 + yy) * HW0 + xx) * 2 + c];
}

__global__ void maxpool2_kernel(const float* __restrict__ x, float* __restrict__ y,
                                int C, int H, int W) {
    const int Ho = H >> 1, Wo = W >> 1;
    int i = blockIdx.x * blockDim.x + threadIdx.x;
    if (i >= BATCH * C * Ho * Wo) return;
    int wo = i % Wo; int t = i / Wo;
    int ho = t % Ho; t /= Ho;
    int c  = t % C;  int b = t / C;
    const float* p = x + (((size_t)b * C + c) * H + ho * 2) * W + wo * 2;
    y[i] = fmaxf(fmaxf(p[0], p[1]), fmaxf(p[W], p[W + 1]));
}

// out (B,16,2h,2w): ch 0..7 = bilinear x2 (align_corners) of low (B,8,h,w);
//                   ch 8..15 = skip (B,8,2h,2w)
__global__ void upsample_concat(const float* __restrict__ low,
                                const float* __restrict__ skip,
                                float* __restrict__ out, int h, int w) {
    const int H = 2 * h, W = 2 * w;
    int i = blockIdx.x * blockDim.x + threadIdx.x;
    if (i >= BATCH * 16 * H * W) return;
    int xo = i % W; int t = i / W;
    int yo = t % H; t /= H;
    int c  = t % 16; int b = t / 16;
    float v;
    if (c < 8) {
        float cy = (float)yo * (float)(h - 1) / (float)(H - 1);
        float cx = (float)xo * (float)(w - 1) / (float)(W - 1);
        int y0i = (int)floorf(cy), x0i = (int)floorf(cx);
        int y1i = min(y0i + 1, h - 1), x1i = min(x0i + 1, w - 1);
        float ty = cy - (float)y0i, tx = cx - (float)x0i;
        const float* p = low + ((size_t)b * 8 + c) * h * w;
        float v00 = p[y0i * w + x0i], v01 = p[y0i * w + x1i];
        float v10 = p[y1i * w + x0i], v11 = p[y1i * w + x1i];
        v = v00 * (1.0f - ty) * (1.0f - tx) + v01 * (1.0f - ty) * tx
          + v10 * ty * (1.0f - tx)          + v11 * ty * tx;
    } else {
        v = skip[(((size_t)b * 8 + (c - 8)) * H + yo) * W + xo];
    }
    out[i] = v;
}

// out (NHWC) = zf + NCHW->NHWC(x)
__global__ void final_out_kernel(const float* __restrict__ zf,
                                 const float* __restrict__ xc,
                                 float* __restrict__ out) {
    int i = blockIdx.x * blockDim.x + threadIdx.x;
    if (i >= BATCH * HW0 * HW0 * 2) return;
    int c  = i & 1;  int t = i >> 1;
    int xx = t % HW0; t /= HW0;
    int yy = t % HW0; int b = t / HW0;
    out[i] = zf[i] + xc[(((size_t)b * 2 + c) * HW0 + yy) * HW0 + xx];
}

// ---------------------------------------------------------------------------
// Host orchestration
// ---------------------------------------------------------------------------
extern "C" void kernel_launch(void* const* d_in, const int* in_sizes, int n_in,
                              void* d_out, int out_size, void* d_ws, size_t ws_size,
                              hipStream_t stream) {
    (void)in_sizes; (void)n_in; (void)out_size; (void)ws_size;
    const float* zf    = (const float*)d_in[0];
    // d_in[1] == y : unused by the reference
    const float* hyper = (const float*)d_in[2];
    const float* hw1 = (const float*)d_in[3];  const float* hb1 = (const float*)d_in[4];
    const float* hw2 = (const float*)d_in[5];  const float* hb2 = (const float*)d_in[6];
    const float* hw3 = (const float*)d_in[7];  const float* hb3 = (const float*)d_in[8];
    const float* hw4 = (const float*)d_in[9];  const float* hb4 = (const float*)d_in[10];
    const float* hw5 = (const float*)d_in[11]; const float* hb5 = (const float*)d_in[12];
    const float* hwo = (const float*)d_in[13]; const float* hbo = (const float*)d_in[14];

    // Workspace layout (floats). Activation buffers padded to [16][Kpad].
    float* F = (float*)d_ws;
    float* hA      = F;                       // 16 x 10016
    float* hB      = F + 160256;              // 16 x 10016
    float* h1      = F + 320512;              // 16 x 128
    float* h2      = F + 322560;              // 16 x 1024
    float* flat    = F + 338944;              // 8 x 9490 (compact)
    float* partial = F + 414864;              // 8 splits x 8 x 10000
    float* bufA    = F + 1054864;             // up to 8 x 16 x 256 x 256
    float* bufB    = bufA + 8388608;
    float* c1      = bufB + 8388608;          // 8 x 8 x 256^2
    float* c2      = c1 + 4194304;            // 8 x 8 x 128^2
    float* c3      = c2 + 1048576;            // 8 x 8 x 64^2
    float* c4      = c3 + 262144;             // 8 x 8 x 32^2

    float* out    = (float*)d_out;            // (8,256,256,2) NHWC
    float* out_l1 = out + BATCH * HW0 * HW0 * 2;

    // ---- Hypernetwork MLP (bandwidth-bound skinny GEMMs) ----
    h1_kernel<<<(16 * 128 + 255) / 256, 256, 0, stream>>>(hyper, hw1, hb1, h1);

    auto gemm_layer = [&](const float* hin, int lda, const float* Wt,
                          const float* bias, float* outp, int K, int N,
                          int ldo, int rows, int nsplit, int act) {
        int  ntiles = (N + 15) / 16;
        dim3 grid((ntiles + 3) / 4, nsplit, 1);
        hyper_gemm_wmma<<<grid, 128, 0, stream>>>(hin, lda, Wt, partial, K, N);
        int total = rows * ldo;
        bias_act_kernel<<<(total + 255) / 256, 256, 0, stream>>>(
            partial, nsplit, bias, outp, N, ldo, rows, act);
    };
    //           hin  lda    W    bias  out    K      N      ldo    rows split act
    gemm_layer(h1,   128, hw2, hb2, h2,     100,  1000,  1024, 16, 1, 1);
    gemm_layer(h2,  1024, hw3, hb3, hA,    1000, 10000, 10016, 16, 4, 1);
    gemm_layer(hA, 10016, hw4, hb4, hB,   10000, 10000, 10016, 16, 8, 1);
    gemm_layer(hB, 10016, hw5, hb5, hA,   10000, 10000, 10016, 16, 8, 1);
    gemm_layer(hA, 10016, hwo, hbo, flat, 10000, FLATLEN, FLATLEN, 8, 8, 0);

    l1_kernel<<<BATCH, 256, 0, stream>>>(flat, out_l1);

    // ---- U-Net with per-sample generated weights ----
    static const int W_OFF[15] = {0, 144, 720, 1296, 1872, 2448, 3024, 3600,
                                  4176, 5328, 5904, 7056, 7632, 8784, 9360};
    auto B_OFF = [](int li) { return 9376 + 8 * li; };     // li<14; li==14 -> 9488

    auto conv = [&](const float* xin, float* yout, int Cin, int Cout, int hw,
                    int ks, int li, int relu) {
        dim3 grid((hw + 15) / 16, hw, BATCH);
        if (ks == 3 && Cin == 2)
            conv_wmma<3, 2><<<grid, 32, 0, stream>>>(xin, flat, yout, Cout, hw, hw,
                                                     W_OFF[li], B_OFF(li), relu);
        else if (ks == 3 && Cin == 8)
            conv_wmma<3, 8><<<grid, 32, 0, stream>>>(xin, flat, yout, Cout, hw, hw,
                                                     W_OFF[li], B_OFF(li), relu);
        else if (ks == 3 && Cin == 16)
            conv_wmma<3, 16><<<grid, 32, 0, stream>>>(xin, flat, yout, Cout, hw, hw,
                                                      W_OFF[li], B_OFF(li), relu);
        else
            conv_wmma<1, 8><<<grid, 32, 0, stream>>>(xin, flat, yout, Cout, hw, hw,
                                                     W_OFF[li], B_OFF(li), relu);
    };
    auto pool = [&](const float* xin, float* yout, int hw) {
        int total = BATCH * 8 * (hw / 2) * (hw / 2);
        maxpool2_kernel<<<(total + 255) / 256, 256, 0, stream>>>(xin, yout, 8, hw, hw);
    };
    auto upcat = [&](const float* low, const float* skip, float* outp, int hw) {
        int total = BATCH * 16 * (2 * hw) * (2 * hw);
        upsample_concat<<<(total + 255) / 256, 256, 0, stream>>>(low, skip, outp, hw, hw);
    };

    {
        int total = BATCH * 2 * HW0 * HW0;
        nhwc_to_nchw<<<(total + 255) / 256, 256, 0, stream>>>(zf, bufA);
    }
    conv(bufA, bufB, 2, 8, 256, 3, 0, 1);      // down0
    conv(bufB, c1,   8, 8, 256, 3, 1, 1);      // conv1 (skip)
    pool(c1, bufA, 256);
    conv(bufA, bufB, 8, 8, 128, 3, 2, 1);
    conv(bufB, c2,   8, 8, 128, 3, 3, 1);      // conv2 (skip)
    pool(c2, bufA, 128);
    conv(bufA, bufB, 8, 8, 64, 3, 4, 1);
    conv(bufB, c3,   8, 8, 64, 3, 5, 1);       // conv3 (skip)
    pool(c3, bufA, 64);
    conv(bufA, bufB, 8, 8, 32, 3, 6, 1);
    conv(bufB, c4,   8, 8, 32, 3, 7, 1);       // conv4

    upcat(c4, c3, bufA, 32);                   // (8,16,64,64)
    conv(bufA, bufB, 16, 8, 64, 3, 8, 1);
    conv(bufB, bufA,  8, 8, 64, 3, 9, 1);
    upcat(bufA, c2, bufB, 64);                 // (8,16,128,128)
    conv(bufB, bufA, 16, 8, 128, 3, 10, 1);
    conv(bufA, bufB,  8, 8, 128, 3, 11, 1);
    upcat(bufB, c1, bufA, 128);                // (8,16,256,256)
    conv(bufA, bufB, 16, 8, 256, 3, 12, 1);
    conv(bufB, bufA,  8, 8, 256, 3, 13, 1);
    conv(bufA, bufB,  8, 2, 256, 1, 14, 0);    // 1x1, no relu

    {
        int total = BATCH * HW0 * HW0 * 2;
        final_out_kernel<<<(total + 255) / 256, 256, 0, stream>>>(zf, bufB, out);
    }
}
